// GSA_83803401880205
// MI455X (gfx1250) — matrix operand order repaired and checked by reading, not compiled, not run on hardware
//
#include <hip/hip_runtime.h>
#include <hip/hip_bf16.h>
#include <stdint.h>

#define EMBED 256
#define HEADS 8
#define HD    32
#define BATCH 4
#define HH    48
#define WW    48
#define LTOK  (HH*WW)          /* 2304 */
#define NTOK  (BATCH*LTOK)     /* 9216 */
#define KCHUNKS (LTOK/32)      /* 72  */
#define SCALE_K 0.17677669529663687f

typedef __attribute__((ext_vector_type(16))) _Float16 v16h;
typedef __attribute__((ext_vector_type(8)))  _Float16 v8h;
typedef __attribute__((ext_vector_type(8)))  float    v8f;
typedef __attribute__((ext_vector_type(4)))  unsigned int u32x4;
typedef __attribute__((ext_vector_type(8)))  int      i32x8;
typedef __attribute__((ext_vector_type(4)))  int      i32x4;

#define WMMA_F16(A,B,C) __builtin_amdgcn_wmma_f32_16x16x32_f16(false,(A),false,(B),(short)0,(C),false,false)

#if defined(__gfx1250__) && __has_builtin(__builtin_amdgcn_tensor_load_to_lds) && __has_builtin(__builtin_amdgcn_s_wait_tensorcnt)
#define USE_TDM 1
#else
#define USE_TDM 0
#endif

// ---------------------------------------------------------------------------
// Fragment loaders.
// A (16x32 f16): lane m=lane&15 holds row m; K halves: VGPR pairs cover
//   K = kb..kb+7 and 16+kb..16+kb+7 with kb = (lane>=16)*8  (ISA 7.12.2)
// B (32x16 f16): lane n=lane&15 holds column n; same K split as A:
//   K = kb..kb+7 and 16+kb..16+kb+7.  We store all B operands pre-transposed
//   (BT[n][k] contiguous in k) so these are two 16B loads per lane.
// ---------------------------------------------------------------------------
__device__ __forceinline__ v16h load_A(const _Float16* base, int rowStride, int lane) {
  const int m  = lane & 15;
  const int kb = (lane & 16) ? 8 : 0;
  const _Float16* p = base + (size_t)m * rowStride;
  v8h lo = *(const v8h*)(p + kb);
  v8h hi = *(const v8h*)(p + 16 + kb);
  v16h a;
#pragma unroll
  for (int i = 0; i < 8; ++i) { a[i] = lo[i]; a[i + 8] = hi[i]; }
  return a;
}

__device__ __forceinline__ v16h load_BT(const _Float16* baseT, int rowStride, int lane) {
  const int n    = lane & 15;
  const int koff = (lane & 16) ? 8 : 0;
  const _Float16* p = baseT + (size_t)n * rowStride + koff;
  v8h lo = *(const v8h*)(p);
  v8h hi = *(const v8h*)(p + 16);
  v16h b;
#pragma unroll
  for (int i = 0; i < 8; ++i) { b[i] = lo[i]; b[i + 8] = hi[i]; }
  return b;
}

// ---------------------------------------------------------------------------
// TDM: async 16x32 fp32 tile load (mask tile) from global into LDS.
// D# built per CDNA5 ISA (2D tensor, groups 2/3 zero).
// ---------------------------------------------------------------------------
#if USE_TDM
__device__ __forceinline__ void tdm_load_mask_tile(const float* gsrc, void* lds_dst) {
  const unsigned           ldsOff = (unsigned)(size_t)lds_dst;      // flat LDS addr low 32 = offset
  const unsigned long long ga     = (unsigned long long)(size_t)gsrc;
  u32x4 g0;
  g0[0] = 1u;                                         // count=1 valid user descriptor
  g0[1] = ldsOff;                                     // lds_addr
  g0[2] = (unsigned)(ga & 0xFFFFFFFFu);               // global_addr[31:0]
  g0[3] = (unsigned)((ga >> 32) & 0x01FFFFFFu)        // global_addr[56:32]
        | (2u << 30);                                 // type = 2 ("image")
  i32x8 g1;
  g1[0] = (int)(2u << 16);                            // data_size=4B, no multicast/pad/iterate
  g1[1] = (int)((LTOK & 0xFFFFu) << 16);              // tensor_dim0 low16 (bits 63:48)
  g1[2] = (int)((LTOK & 0xFFFFu) << 16);              // dim0 hi=0 | tensor_dim1 low16 (95:80)
  g1[3] = (int)(32u << 16);                           // dim1 hi=0 | tile_dim0=32 (127:112)
  g1[4] = (int)16;                                    // tile_dim1=16, tile_dim2=0
  g1[5] = (int)LTOK;                                  // tensor_dim0_stride low32
  g1[6] = 0;                                          // stride0 hi16 | stride1 low16 (2D: unused)
  g1[7] = 0;
  i32x4 gz = {0, 0, 0, 0};
#if __clang_major__ >= 23
  i32x8 gz8 = {0, 0, 0, 0, 0, 0, 0, 0};
  __builtin_amdgcn_tensor_load_to_lds(g0, g1, gz, gz, gz8, 0);
#else
  __builtin_amdgcn_tensor_load_to_lds(g0, g1, gz, gz, 0);
#endif
}
#endif

// ---------------------------------------------------------------------------
// Kernel 0a: fp32 -> f16 convert
// ---------------------------------------------------------------------------
__global__ void cvt_f16_kernel(const float* __restrict__ s, _Float16* __restrict__ d, int n) {
  int i = blockIdx.x * blockDim.x + threadIdx.x;
  if (i < n) d[i] = (_Float16)s[i];
}

// Kernel 0b: 256x256 fp32 W -> f16 W^T (BT[n][k] = W[k][n])
__global__ void transpose_f16_kernel(const float* __restrict__ W, _Float16* __restrict__ WT) {
  int idx = blockIdx.x * blockDim.x + threadIdx.x;   // 65536
  int n = idx >> 8, k = idx & 255;
  WT[idx] = (_Float16)W[k * 256 + n];
}

// ---------------------------------------------------------------------------
// Kernel 1: fused QKV projection + bias + k-scale + RoPE.
// 4 waves/block; wave -> one 16x16 output tile per projection.
// ---------------------------------------------------------------------------
__global__ __launch_bounds__(128) void qkv_rope_kernel(
    const _Float16* __restrict__ xh,
    const _Float16* __restrict__ WqT, const _Float16* __restrict__ WkT,
    const _Float16* __restrict__ WvT,
    const float* __restrict__ bq, const float* __restrict__ bk, const float* __restrict__ bv,
    const float* __restrict__ sinT, const float* __restrict__ cosT,
    _Float16* __restrict__ Qh, _Float16* __restrict__ Kh, _Float16* __restrict__ Vt,
    float* __restrict__ vf32) {
  const int lane = threadIdx.x & 31;
  const int wv   = threadIdx.x >> 5;
  const int m0   = blockIdx.x * 16;
  const int n0   = (blockIdx.y * 4 + wv) * 16;

  v8f cq = {}; v8f ck = {}; v8f cv = {};
#pragma unroll
  for (int kk = 0; kk < 8; ++kk) {
    const int k0 = kk * 32;
    v16h a   = load_A (xh  + (size_t)m0 * EMBED + k0, EMBED, lane);
    v16h bqf = load_BT(WqT + (size_t)n0 * EMBED + k0, EMBED, lane);
    v16h bkf = load_BT(WkT + (size_t)n0 * EMBED + k0, EMBED, lane);
    v16h bvf = load_BT(WvT + (size_t)n0 * EMBED + k0, EMBED, lane);
    cq = WMMA_F16(a, bqf, cq);
    ck = WMMA_F16(a, bkf, ck);
    cv = WMMA_F16(a, bvf, cv);
  }
  const int n     = lane & 15;
  const int c     = n0 + n;
  const int head  = c >> 5;
  const int d     = c & 31;
  const int mhalf = (lane & 16) ? 8 : 0;
  const float bqv = bq[c], bkv = bk[c], bvv = bv[c];
  const float sgn = (c & 1) ? 1.f : -1.f;   // rot[2i]=-x[2i+1], rot[2i+1]=x[2i]
#pragma unroll
  for (int r = 0; r < 8; ++r) {
    const int token = m0 + r + mhalf;
    const int b     = token / LTOK;
    const int pos   = token - b * LTOK;
    const float qv  = cq[r] + bqv;
    const float kv  = (ck[r] + bkv) * SCALE_K;
    const float vv  = cv[r] + bvv;
    const float qp  = __shfl_xor(qv, 1, 32);   // partner channel c^1 lives in lane^1
    const float kp  = __shfl_xor(kv, 1, 32);
    const float sv  = sinT[(size_t)pos * HD + d];
    const float cvl = cosT[(size_t)pos * HD + d];
    const float qo  = qv * cvl + sgn * qp * sv;
    const float ko  = kv * cvl + sgn * kp * sv;
    const size_t bh = (size_t)b * HEADS + head;
    Qh[(bh * LTOK + pos) * HD + d] = (_Float16)qo;
    Kh[(bh * LTOK + pos) * HD + d] = (_Float16)ko;
    Vt[(bh * HD + d) * LTOK + pos] = (_Float16)vv;          // V^T for PV B-fragments
    vf32[(size_t)token * EMBED + c] = vv;                    // fp32 V for LePE conv
  }
}

// ---------------------------------------------------------------------------
// Kernel 2: flash attention.  4 waves/block, wave -> one 16-query tile of one
// (b,h).  Mask tiles streamed via TDM into LDS; P bounced through LDS.
// ---------------------------------------------------------------------------
__global__ __launch_bounds__(128) void attn_kernel(
    const _Float16* __restrict__ Qh, const _Float16* __restrict__ Kh,
    const _Float16* __restrict__ Vt, const float* __restrict__ mask,
    float* __restrict__ attO) {
  __shared__ __align__(16) float    maskT[4][16 * 32];
  __shared__ __align__(16) _Float16 pT[4][16 * 32];
  const int lane = threadIdx.x & 31;
  const int wv   = threadIdx.x >> 5;
  const int q0   = (blockIdx.x * 4 + wv) * 16;
  const int h    = blockIdx.y;
  const int b    = blockIdx.z;
  const size_t bh = (size_t)b * HEADS + h;

  const v16h Aq = load_A(Qh + (bh * LTOK + q0) * HD, HD, lane);
  v8f O0 = {}; v8f O1 = {};
  float mrow[8], lrow[8];
#pragma unroll
  for (int r = 0; r < 8; ++r) { mrow[r] = -1e30f; lrow[r] = 0.f; }
  const int n     = lane & 15;
  const int mhalf = (lane & 16) ? 8 : 0;
  const float* mbase = mask + ((size_t)h * LTOK + q0) * LTOK;

  for (int j = 0; j < KCHUNKS; ++j) {
    const int k0 = j * 32;
    __syncthreads();                           // tiles from prev iter fully consumed
#if USE_TDM
    tdm_load_mask_tile(mbase + k0, &maskT[wv][0]);
#else
#pragma unroll
    for (int it = 0; it < 16; ++it)
      maskT[wv][it * 32 + lane] = mbase[(size_t)it * LTOK + k0 + lane];
#endif
    // S = Q K^T for 32 keys (two 16x16 tiles), overlapped with the mask DMA
    v16h Bk0 = load_BT(Kh + (bh * LTOK + k0) * HD,      HD, lane);
    v16h Bk1 = load_BT(Kh + (bh * LTOK + k0 + 16) * HD, HD, lane);
    v8f S0 = {}; v8f S1 = {};
    S0 = WMMA_F16(Aq, Bk0, S0);
    S1 = WMMA_F16(Aq, Bk1, S1);
#if USE_TDM
    __builtin_amdgcn_s_wait_tensorcnt(0);
#endif
    __syncthreads();                           // mask tile visible
#pragma unroll
    for (int r = 0; r < 8; ++r) {
      float s0 = S0[r] + maskT[wv][(r + mhalf) * 32 + n];
      float s1 = S1[r] + maskT[wv][(r + mhalf) * 32 + 16 + n];
      float rm = fmaxf(s0, s1);
#pragma unroll
      for (int mm = 1; mm < 16; mm <<= 1) rm = fmaxf(rm, __shfl_xor(rm, mm, 32));
      const float newm = fmaxf(mrow[r], rm);
      const float sc   = __expf(mrow[r] - newm);
      const float p0   = __expf(s0 - newm);
      const float p1   = __expf(s1 - newm);
      float rs = p0 + p1;
#pragma unroll
      for (int mm = 1; mm < 16; mm <<= 1) rs += __shfl_xor(rs, mm, 32);
      lrow[r] = lrow[r] * sc + rs;
      mrow[r] = newm;
      O0[r] *= sc; O1[r] *= sc;
      pT[wv][(r + mhalf) * 32 + n]      = (_Float16)p0;
      pT[wv][(r + mhalf) * 32 + 16 + n] = (_Float16)p1;
    }
    __syncthreads();                           // P tile visible
    v16h Ap  = load_A(&pT[wv][0], 32, lane);
    v16h Bv0 = load_BT(Vt + (bh * HD +  0) * LTOK + k0, LTOK, lane);
    v16h Bv1 = load_BT(Vt + (bh * HD + 16) * LTOK + k0, LTOK, lane);
    O0 = WMMA_F16(Ap, Bv0, O0);
    O1 = WMMA_F16(Ap, Bv1, O1);
  }
#pragma unroll
  for (int r = 0; r < 8; ++r) {
    const float inv = 1.0f / lrow[r];
    const int qrow  = q0 + r + mhalf;
    attO[((size_t)b * LTOK + qrow) * EMBED + h * HD + n]      = O0[r] * inv;
    attO[((size_t)b * LTOK + qrow) * EMBED + h * HD + 16 + n] = O1[r] * inv;
  }
}

// ---------------------------------------------------------------------------
// Kernel 3: LePE depthwise 5x5 conv + residual add + f16 convert
// ---------------------------------------------------------------------------
__global__ void lepe_add_kernel(const float* __restrict__ vf32, const float* __restrict__ attO,
                                const float* __restrict__ lw, const float* __restrict__ lb,
                                _Float16* __restrict__ zh) {
  const int idx = blockIdx.x * blockDim.x + threadIdx.x;
  if (idx >= NTOK * EMBED) return;
  const int c     = idx & (EMBED - 1);
  const int token = idx >> 8;
  const int b     = token / LTOK;
  const int pos   = token - b * LTOK;
  const int y     = pos / WW, x = pos - y * WW;
  float acc = lb[c];
#pragma unroll
  for (int dy = 0; dy < 5; ++dy) {
    const int yy = y + dy - 2;
    if (yy < 0 || yy >= HH) continue;
#pragma unroll
    for (int dx = 0; dx < 5; ++dx) {
      const int xx = x + dx - 2;
      if (xx < 0 || xx >= WW) continue;
      acc += vf32[(((size_t)b * LTOK) + yy * WW + xx) * EMBED + c] * lw[(dy * 5 + dx) * EMBED + c];
    }
  }
  zh[idx] = (_Float16)(attO[idx] + acc);
}

// ---------------------------------------------------------------------------
// Kernel 4: output projection  out = z @ Wout + bout  (fp32 out)
// ---------------------------------------------------------------------------
__global__ __launch_bounds__(128) void out_proj_kernel(
    const _Float16* __restrict__ zh, const _Float16* __restrict__ WoT,
    const float* __restrict__ bout, float* __restrict__ out) {
  const int lane = threadIdx.x & 31;
  const int wv   = threadIdx.x >> 5;
  const int m0   = blockIdx.x * 16;
  const int n0   = (blockIdx.y * 4 + wv) * 16;
  v8f acc = {};
#pragma unroll
  for (int kk = 0; kk < 8; ++kk) {
    const int k0 = kk * 32;
    v16h a  = load_A (zh  + (size_t)m0 * EMBED + k0, EMBED, lane);
    v16h bf = load_BT(WoT + (size_t)n0 * EMBED + k0, EMBED, lane);
    acc = WMMA_F16(a, bf, acc);
  }
  const int n     = lane & 15;
  const int c     = n0 + n;
  const int mhalf = (lane & 16) ? 8 : 0;
  const float bo  = bout[c];
#pragma unroll
  for (int r = 0; r < 8; ++r) {
    const int token = m0 + r + mhalf;
    out[(size_t)token * EMBED + c] = acc[r] + bo;
  }
}

// ---------------------------------------------------------------------------
extern "C" void kernel_launch(void* const* d_in, const int* in_sizes, int n_in,
                              void* d_out, int out_size, void* d_ws, size_t ws_size,
                              hipStream_t stream) {
  const float* x     = (const float*)d_in[0];
  const float* sinT  = (const float*)d_in[1];
  const float* cosT  = (const float*)d_in[2];
  const float* mask  = (const float*)d_in[3];
  const float* Wq    = (const float*)d_in[4];
  const float* bq    = (const float*)d_in[5];
  const float* Wk    = (const float*)d_in[6];
  const float* bk    = (const float*)d_in[7];
  const float* Wv    = (const float*)d_in[8];
  const float* bv    = (const float*)d_in[9];
  const float* lw    = (const float*)d_in[10];
  const float* lb    = (const float*)d_in[11];
  const float* Wout  = (const float*)d_in[12];
  const float* bout  = (const float*)d_in[13];
  float* out = (float*)d_out;

  char* base = (char*)d_ws;
  size_t off = 0;
  auto sub = [&](size_t bytes) -> char* {
    char* p = base + off;
    off += (bytes + 255) & ~(size_t)255;
    return p;
  };
  _Float16* xh   = (_Float16*)sub((size_t)NTOK * EMBED * 2);
  _Float16* WqT  = (_Float16*)sub(65536 * 2);
  _Float16* WkT  = (_Float16*)sub(65536 * 2);
  _Float16* WvT  = (_Float16*)sub(65536 * 2);
  _Float16* WoT  = (_Float16*)sub(65536 * 2);
  _Float16* Qh   = (_Float16*)sub((size_t)NTOK * EMBED * 2);
  _Float16* Kh   = (_Float16*)sub((size_t)NTOK * EMBED * 2);
  _Float16* Vt   = (_Float16*)sub((size_t)NTOK * EMBED * 2);
  float*    vf32 = (float*)   sub((size_t)NTOK * EMBED * 4);
  float*    attO = (float*)   sub((size_t)NTOK * EMBED * 4);
  _Float16* zh   = (_Float16*)sub((size_t)NTOK * EMBED * 2);

  const int nElem = NTOK * EMBED;
  cvt_f16_kernel<<<nElem / 256, 256, 0, stream>>>(x, xh, nElem);
  transpose_f16_kernel<<<256, 256, 0, stream>>>(Wq, WqT);
  transpose_f16_kernel<<<256, 256, 0, stream>>>(Wk, WkT);
  transpose_f16_kernel<<<256, 256, 0, stream>>>(Wv, WvT);
  transpose_f16_kernel<<<256, 256, 0, stream>>>(Wout, WoT);

  qkv_rope_kernel<<<dim3(NTOK / 16, 4), 128, 0, stream>>>(
      xh, WqT, WkT, WvT, bq, bk, bv, sinT, cosT, Qh, Kh, Vt, vf32);

  attn_kernel<<<dim3(LTOK / 64, HEADS, BATCH), 128, 0, stream>>>(Qh, Kh, Vt, mask, attO);

  lepe_add_kernel<<<nElem / 256, 256, 0, stream>>>(vf32, attO, lw, lb, zh);

  out_proj_kernel<<<dim3(NTOK / 16, 4), 128, 0, stream>>>(zh, WoT, bout, out);
}
